// MesoBranched_59459527246614
// MI455X (gfx1250) — matrix operand order repaired and checked by reading, not compiled, not run on hardware
//
#include <hip/hip_runtime.h>

// ---- problem constants (match reference) ----
#define NNODES 50000
#define NEDGES 500000
#define FIN    256
#define HDIM   128
#define TDIM   2
#define GDIM   8
#define BN_EPS 1e-5f

// ---- WMMA fragment types (CDNA5 / gfx1250, wave32) ----
typedef __attribute__((ext_vector_type(16))) __bf16        v16bf;
typedef __attribute__((ext_vector_type(8)))  float         v8f;
typedef __attribute__((ext_vector_type(4)))  unsigned int  u32x4;

union BfFrag { v16bf v; u32x4 q[2]; };

__device__ __forceinline__ unsigned short f2bf(float f) {
  union { __bf16 b; unsigned short u; } c; c.b = (__bf16)f; return c.u;
}
__device__ __forceinline__ float bf2f(unsigned short u) {
  union { unsigned short u; __bf16 b; } c; c.u = u; return (float)c.b;
}

__device__ __forceinline__ void atomicMaxF(float* addr, float val) {
  unsigned int* ua = (unsigned int*)addr;
  unsigned int old = *ua;
  while (true) {
    float f = __uint_as_float(old);
    if (f >= val) break;
    unsigned int assumed = old;
    old = atomicCAS(ua, assumed, __float_as_uint(val));
    if (old == assumed) break;
  }
}

// =====================================================================
// GEMM: Hout[M,128] = A[M,K](bf16) @ W[K,128](bf16, stored Wt[128][K]) + bias
// Also accumulates per-column sum / sumsq of valid rows into ssum/ssq
// (training-mode BatchNorm statistics).
//
// Block: 256 threads = 8 waves; 128 rows x 128 cols per block.
// Wt is staged in LDS (padded pitch to spread banks); B fragments come from
// ds_load_b128. A fragments are double-buffered from global one k-step ahead.
// =====================================================================
#define WPAD 8  // halfwords of row padding in LDS (16B, keeps b128 alignment)

__global__ __launch_bounds__(256) void mb_gemm_bf16(
    const unsigned short* __restrict__ A, const unsigned short* __restrict__ Wt,
    const float* __restrict__ bias, float* __restrict__ Hout,
    float* __restrict__ ssum, float* __restrict__ ssq,
    long long M, int K) {
  __shared__ unsigned short sW[HDIM * (FIN + WPAD)];  // 67584 B (max K=256)
  __shared__ float lsum[HDIM];
  __shared__ float lsq[HDIM];

  const int tid = threadIdx.x;
  const int pitch = K + WPAD;

  // ---- cooperative stage of Wt[128][K] into LDS (b128 chunks) ----
  {
    const int chunksPerRow = K >> 3;  // 8 halfwords per u32x4
    const int totalChunks = HDIM * chunksPerRow;
    for (int i = tid; i < totalChunks; i += 256) {
      const int row = i / chunksPerRow;
      const int ch = i - row * chunksPerRow;
      *(u32x4*)(&sW[row * pitch + ch * 8]) =
          *(const u32x4*)(Wt + (size_t)row * K + ch * 8);
    }
  }
  if (tid < HDIM) { lsum[tid] = 0.f; lsq[tid] = 0.f; }
  __syncthreads();

  const int wave = tid >> 5;
  const int lane = tid & 31;
  const int half = lane >> 4;   // 0: lanes 0-15, 1: lanes 16-31
  const int l16  = lane & 15;

  const long long rowBase = (long long)blockIdx.x * 128 + (long long)wave * 16;
  const long long aRow    = rowBase + l16;   // A-matrix row held by this lane
  const bool aValid       = aRow < M;
  // A fragment (16-bit A 16x32 layout): halfwords 0..7 -> K=k0+half*8+h,
  // halfwords 8..15 -> K=k0+16+half*8+(h-8); both groups 16B contiguous.
  const unsigned short* aPtr =
      A + (size_t)(aValid ? aRow : 0) * (size_t)K + half * 8;

  const v8f z8 = {0.f, 0.f, 0.f, 0.f, 0.f, 0.f, 0.f, 0.f};
  v8f acc[8];
  #pragma unroll
  for (int t = 0; t < 8; ++t) acc[t] = z8;

  const u32x4 zq = {0u, 0u, 0u, 0u};

  // preload A fragment for k0 = 0
  BfFrag a, an;
  a.q[0] = *(const u32x4*)(aPtr);
  a.q[1] = *(const u32x4*)(aPtr + 16);
  if (!aValid) { a.q[0] = zq; a.q[1] = zq; }

  for (int k0 = 0; k0 < K; k0 += 32) {
    const bool more = (k0 + 32) < K;
    if (more) {
      // double-buffer next A fragment; prefetch one step beyond that
      an.q[0] = *(const u32x4*)(aPtr + k0 + 32);
      an.q[1] = *(const u32x4*)(aPtr + k0 + 48);
      __builtin_prefetch(aPtr + k0 + 64, 0, 3);
    }
    // B fragment (32x16): lane holds column n = t*16 + l16,
    // halfwords 0..7 -> K = k0 + half*8 + h, 8..15 -> K = k0 + 16 + half*8 + (h-8)
    const int wbase = k0 + half * 8;
    #pragma unroll
    for (int t = 0; t < 8; ++t) {
      BfFrag b;
      const int wrow = (t * 16 + l16) * pitch + wbase;
      b.q[0] = *(const u32x4*)(&sW[wrow]);
      b.q[1] = *(const u32x4*)(&sW[wrow + 16]);
      acc[t] = __builtin_amdgcn_wmma_f32_16x16x32_bf16(
          false, a.v, false, b.v, (short)0, acc[t], false, false);
    }
    if (more) {
      if (!aValid) { an.q[0] = zq; an.q[1] = zq; }
      a = an;
    }
  }

  // D layout: VGPR r holds element (M = r + half*8, N = l16) of each tile
  #pragma unroll
  for (int t = 0; t < 8; ++t) {
    const int n = t * 16 + l16;
    const float bv = bias[n];
    float s0 = 0.f, s1 = 0.f;
    #pragma unroll
    for (int r = 0; r < 8; ++r) {
      const long long row = rowBase + half * 8 + r;
      if (row < M) {
        const float h = acc[t][r] + bv;
        Hout[(size_t)row * HDIM + n] = h;
        s0 += h;
        s1 += h * h;
      }
    }
    // lane L and L^16 hold the same column n over complementary row halves
    s0 += __shfl_xor(s0, 16, 32);
    s1 += __shfl_xor(s1, 16, 32);
    if (half == 0) {
      atomicAdd(&lsum[n], s0);
      atomicAdd(&lsq[n], s1);
    }
  }
  __syncthreads();
  if (tid < HDIM) {
    atomicAdd(&ssum[tid], lsum[tid]);
    atomicAdd(&ssq[tid], lsq[tid]);
  }
}

// ---- fp32 -> bf16 elementwise convert ----
__global__ void mb_cvt_bf16(const float* __restrict__ in,
                            unsigned short* __restrict__ out, long long n) {
  long long i = (long long)blockIdx.x * blockDim.x + threadIdx.x;
  const long long stride = (long long)gridDim.x * blockDim.x;
  for (; i < n; i += stride) out[i] = f2bf(in[i]);
}

// ---- weight convert+transpose: W[K][128] fp32 -> Wt[128][K] bf16 ----
__global__ void mb_cvt_w(const float* __restrict__ W,
                         unsigned short* __restrict__ Wt, int K) {
  const int total = HDIM * K;
  for (int i = blockIdx.x * blockDim.x + threadIdx.x; i < total;
       i += gridDim.x * blockDim.x) {
    const int n = i / K, k = i % K;
    Wt[i] = f2bf(W[(size_t)k * HDIM + n]);
  }
}

// ---- BN affine params from batch statistics (biased variance) ----
__global__ void mb_bnparams(const float* __restrict__ ssum,
                            const float* __restrict__ ssq,
                            const float* __restrict__ g,
                            const float* __restrict__ be, float invM,
                            float* __restrict__ scale,
                            float* __restrict__ shift) {
  const int c = threadIdx.x;
  const float mean = ssum[c] * invM;
  const float var  = ssq[c] * invM - mean * mean;
  const float s    = g[c] * rsqrtf(var + BN_EPS);
  scale[c] = s;
  shift[c] = be[c] - mean * s;
}

// ---- relu(bn(h)) -> bf16 activations ----
__global__ void mb_bn_relu_bf16(const float* __restrict__ H,
                                const float* __restrict__ scale,
                                const float* __restrict__ shift,
                                unsigned short* __restrict__ out, long long n) {
  long long i = (long long)blockIdx.x * blockDim.x + threadIdx.x;
  const long long stride = (long long)gridDim.x * blockDim.x;
  for (; i < n; i += stride) {
    const int c = (int)(i & (HDIM - 1));
    const float v = fmaxf(fmaf(H[i], scale[c], shift[c]), 0.f);
    out[i] = f2bf(v);
  }
}

// ---- edge_feat[e] = [ x[dst[e]], x[src[e]] - x[dst[e]] ]  (bf16) ----
__global__ void mb_edge_feat(const unsigned short* __restrict__ xb,
                             const int* __restrict__ src,
                             const int* __restrict__ dst,
                             unsigned short* __restrict__ feat, int E) {
  const long long total = (long long)E * HDIM;
  long long i = (long long)blockIdx.x * blockDim.x + threadIdx.x;
  const long long stride = (long long)gridDim.x * blockDim.x;
  for (; i < total; i += stride) {
    const long long e = i >> 7;
    const int c = (int)(i & (HDIM - 1));
    const int d = dst[e], s = src[e];
    const unsigned short xi = xb[(size_t)d * HDIM + c];
    const unsigned short xj = xb[(size_t)s * HDIM + c];
    feat[(size_t)e * (2 * HDIM) + c] = xi;
    feat[(size_t)e * (2 * HDIM) + HDIM + c] = f2bf(bf2f(xj) - bf2f(xi));
  }
}

// ---- msg = relu(bn(h2)); scatter-add into xnext[dst] ----
__global__ void mb_aggregate(const float* __restrict__ H,
                             const float* __restrict__ scale,
                             const float* __restrict__ shift,
                             const int* __restrict__ dst,
                             float* __restrict__ xnext, int E) {
  const long long total = (long long)E * HDIM;
  long long i = (long long)blockIdx.x * blockDim.x + threadIdx.x;
  const long long stride = (long long)gridDim.x * blockDim.x;
  for (; i < total; i += stride) {
    const long long e = i >> 7;
    const int c = (int)(i & (HDIM - 1));
    const float v = fmaxf(fmaf(H[i], scale[c], shift[c]), 0.f);
    atomicAdd(&xnext[(size_t)dst[e] * HDIM + c], v);
  }
}

// ---- mean aggregation: x = xnext / max(cnt,1), stored bf16 ----
__global__ void mb_finalize_x(const float* __restrict__ xnext,
                              const float* __restrict__ cnt,
                              unsigned short* __restrict__ xb) {
  const long long total = (long long)NNODES * HDIM;
  long long i = (long long)blockIdx.x * blockDim.x + threadIdx.x;
  const long long stride = (long long)gridDim.x * blockDim.x;
  for (; i < total; i += stride) {
    const long long n = i >> 7;
    const float d = fmaxf(cnt[n], 1.f);
    xb[i] = f2bf(xnext[i] / d);
  }
}

// ---- per-target-node edge counts ----
__global__ void mb_count(const int* __restrict__ dst, float* __restrict__ cnt,
                         int E) {
  for (int i = blockIdx.x * blockDim.x + threadIdx.x; i < E;
       i += gridDim.x * blockDim.x)
    atomicAdd(&cnt[dst[i]], 1.0f);
}

// ---- Z[n] += x[n] @ lin_W + lin_b   (H=128 -> T=2 GEMV) ----
__global__ void mb_readout(const unsigned short* __restrict__ xb,
                           const float* __restrict__ W,
                           const float* __restrict__ b, float* __restrict__ Z,
                           int N) {
  const int n = blockIdx.x * blockDim.x + threadIdx.x;
  if (n >= N) return;
  float a0 = b[0], a1 = b[1];
  const unsigned short* xr = xb + (size_t)n * HDIM;
  #pragma unroll 8
  for (int c = 0; c < HDIM; ++c) {
    const float x = bf2f(xr[c]);
    a0 = fmaf(x, W[c * 2 + 0], a0);
    a1 = fmaf(x, W[c * 2 + 1], a1);
  }
  Z[(size_t)n * 2 + 0] += a0;
  Z[(size_t)n * 2 + 1] += a1;
}

__global__ void mb_init_pool(float* __restrict__ out) {
  out[threadIdx.x] = -3.402823466e+38f;
}

// ---- segment max pool over graphs ----
__global__ void mb_pool(const float* __restrict__ Z,
                        const int* __restrict__ batch, float* __restrict__ out,
                        int N) {
  for (int n = blockIdx.x * blockDim.x + threadIdx.x; n < N;
       n += gridDim.x * blockDim.x) {
    const int g = batch[n];
    atomicMaxF(&out[g * TDIM + 0], Z[(size_t)n * 2 + 0]);
    atomicMaxF(&out[g * TDIM + 1], Z[(size_t)n * 2 + 1]);
  }
}

// =====================================================================
extern "C" void kernel_launch(void* const* d_in, const int* in_sizes, int n_in,
                              void* d_out, int out_size, void* d_ws,
                              size_t ws_size, hipStream_t stream) {
  (void)in_sizes; (void)n_in; (void)out_size; (void)ws_size;

  const float* xfeat  = (const float*)d_in[0];
  const int*   eidx   = (const int*)d_in[1];
  const int*   src    = eidx;            // edge_index[0]
  const int*   dst    = eidx + NEDGES;   // edge_index[1]
  const int*   batch  = (const int*)d_in[2];
  const float* fh_W1  = (const float*)d_in[3];
  const float* fh_b1  = (const float*)d_in[4];
  const float* fh_g1  = (const float*)d_in[5];
  const float* fh_be1 = (const float*)d_in[6];
  const float* fh_W2  = (const float*)d_in[7];
  const float* fh_b2  = (const float*)d_in[8];
  const float* fh_g2  = (const float*)d_in[9];
  const float* fh_be2 = (const float*)d_in[10];
  const float* ec_W1  = (const float*)d_in[11];
  const float* ec_b1  = (const float*)d_in[12];
  const float* ec_g1  = (const float*)d_in[13];
  const float* ec_be1 = (const float*)d_in[14];
  const float* ec_W2  = (const float*)d_in[15];
  const float* ec_b2  = (const float*)d_in[16];
  const float* ec_g2  = (const float*)d_in[17];
  const float* ec_be2 = (const float*)d_in[18];
  const float* lin_W  = (const float*)d_in[19];
  const float* lin_b  = (const float*)d_in[20];

  float* out = (float*)d_out;        // [G*T] pooled max
  float* Z   = out + GDIM * TDIM;    // [N*T]

  // ---- workspace carve-out ----
  char* w = (char*)d_ws;
  auto alloc = [&](size_t bytes) {
    char* p = w;
    w += (bytes + 255) & ~(size_t)255;
    return p;
  };
  unsigned short* xfeat_bf = (unsigned short*)alloc((size_t)NNODES * FIN * 2);
  unsigned short* x_bf     = (unsigned short*)alloc((size_t)NNODES * HDIM * 2);
  unsigned short* Abuf     = (unsigned short*)alloc((size_t)NEDGES * 2 * HDIM * 2);
  float*          Hbuf     = (float*)alloc((size_t)NEDGES * HDIM * 4);
  float*          xnext    = (float*)alloc((size_t)NNODES * HDIM * 4);
  float*          cnt      = (float*)alloc((size_t)NNODES * 4);
  unsigned short* Wt       = (unsigned short*)alloc((size_t)HDIM * FIN * 2);
  float*          ssum     = (float*)alloc(HDIM * 4);
  float*          ssq      = (float*)alloc(HDIM * 4);
  float*          scale    = (float*)alloc(HDIM * 4);
  float*          shift    = (float*)alloc(HDIM * 4);

  const dim3 blk(256);
  const int EW = 2048;  // grid blocks for elementwise grid-stride kernels

  // ================= first_h MLP on nodes =================
  mb_cvt_bf16<<<EW, blk, 0, stream>>>(xfeat, xfeat_bf, (long long)NNODES * FIN);

  hipMemsetAsync(ssum, 0, HDIM * 4, stream);
  hipMemsetAsync(ssq,  0, HDIM * 4, stream);
  mb_cvt_w<<<64, blk, 0, stream>>>(fh_W1, Wt, FIN);
  mb_gemm_bf16<<<(NNODES + 127) / 128, blk, 0, stream>>>(
      xfeat_bf, Wt, fh_b1, Hbuf, ssum, ssq, (long long)NNODES, FIN);
  mb_bnparams<<<1, HDIM, 0, stream>>>(ssum, ssq, fh_g1, fh_be1,
                                      1.f / (float)NNODES, scale, shift);
  mb_bn_relu_bf16<<<EW, blk, 0, stream>>>(Hbuf, scale, shift, Abuf,
                                          (long long)NNODES * HDIM);

  hipMemsetAsync(ssum, 0, HDIM * 4, stream);
  hipMemsetAsync(ssq,  0, HDIM * 4, stream);
  mb_cvt_w<<<64, blk, 0, stream>>>(fh_W2, Wt, HDIM);
  mb_gemm_bf16<<<(NNODES + 127) / 128, blk, 0, stream>>>(
      Abuf, Wt, fh_b2, Hbuf, ssum, ssq, (long long)NNODES, HDIM);
  mb_bnparams<<<1, HDIM, 0, stream>>>(ssum, ssq, fh_g2, fh_be2,
                                      1.f / (float)NNODES, scale, shift);
  mb_bn_relu_bf16<<<EW, blk, 0, stream>>>(Hbuf, scale, shift, x_bf,
                                          (long long)NNODES * HDIM);

  // ================= readout init (layer 0) =================
  hipMemsetAsync(Z, 0, (size_t)NNODES * TDIM * 4, stream);
  mb_init_pool<<<1, GDIM * TDIM, 0, stream>>>(out);
  mb_readout<<<(NNODES + 255) / 256, blk, 0, stream>>>(x_bf, lin_W, lin_b, Z,
                                                       NNODES);

  // ================= per-node in-degree counts =================
  hipMemsetAsync(cnt, 0, (size_t)NNODES * 4, stream);
  mb_count<<<EW, blk, 0, stream>>>(dst, cnt, NEDGES);

  // ================= 2 EdgeConv layers =================
  for (int l = 0; l < 2; ++l) {
    mb_edge_feat<<<EW, blk, 0, stream>>>(x_bf, src, dst, Abuf, NEDGES);

    hipMemsetAsync(ssum, 0, HDIM * 4, stream);
    hipMemsetAsync(ssq,  0, HDIM * 4, stream);
    mb_cvt_w<<<64, blk, 0, stream>>>(ec_W1 + (size_t)l * 2 * HDIM * HDIM, Wt,
                                     2 * HDIM);
    mb_gemm_bf16<<<(NEDGES + 127) / 128, blk, 0, stream>>>(
        Abuf, Wt, ec_b1 + l * HDIM, Hbuf, ssum, ssq, (long long)NEDGES,
        2 * HDIM);
    mb_bnparams<<<1, HDIM, 0, stream>>>(ssum, ssq, ec_g1 + l * HDIM,
                                        ec_be1 + l * HDIM,
                                        1.f / (float)NEDGES, scale, shift);
    // A2 = relu(bn(h1)) reuses Abuf (edge_feat no longer needed)
    mb_bn_relu_bf16<<<EW, blk, 0, stream>>>(Hbuf, scale, shift, Abuf,
                                            (long long)NEDGES * HDIM);

    hipMemsetAsync(ssum, 0, HDIM * 4, stream);
    hipMemsetAsync(ssq,  0, HDIM * 4, stream);
    mb_cvt_w<<<64, blk, 0, stream>>>(ec_W2 + (size_t)l * HDIM * HDIM, Wt, HDIM);
    mb_gemm_bf16<<<(NEDGES + 127) / 128, blk, 0, stream>>>(
        Abuf, Wt, ec_b2 + l * HDIM, Hbuf, ssum, ssq, (long long)NEDGES, HDIM);
    mb_bnparams<<<1, HDIM, 0, stream>>>(ssum, ssq, ec_g2 + l * HDIM,
                                        ec_be2 + l * HDIM,
                                        1.f / (float)NEDGES, scale, shift);

    hipMemsetAsync(xnext, 0, (size_t)NNODES * HDIM * 4, stream);
    mb_aggregate<<<EW, blk, 0, stream>>>(Hbuf, scale, shift, dst, xnext,
                                         NEDGES);
    mb_finalize_x<<<EW, blk, 0, stream>>>(xnext, cnt, x_bf);
    mb_readout<<<(NNODES + 255) / 256, blk, 0, stream>>>(
        x_bf, lin_W + (size_t)(l + 1) * HDIM * TDIM, lin_b + (l + 1) * TDIM, Z,
        NNODES);
  }

  // ================= global max pool =================
  mb_pool<<<EW, blk, 0, stream>>>(Z, batch, out, NNODES);
}
